// Gru_cond_layer_13932873908771
// MI455X (gfx1250) — matrix operand order repaired and checked by reading, not compiled, not run on hardware
//
#include <hip/hip_runtime.h>
#include <hip/hip_bf16.h>
#include <stdint.h>

// ---------------------------------------------------------------------------
// Problem constants (from the reference)
// ---------------------------------------------------------------------------
constexpr int TT  = 32;     // timesteps
constexpr int BB  = 8;      // batch
constexpr int MM  = 256;    // embedding dim
constexpr int NN  = 256;    // hidden dim
constexpr int DD  = 684;    // context channels
constexpr int DPAD= 704;    // 684 padded to multiple of 32
constexpr int AA  = 512;    // attention dim
constexpr int HH  = 16;
constexpr int WW  = 48;
constexpr int HWX = HH * WW;        // 768
constexpr int HWB = HWX * BB;       // 6144 attention rows
constexpr int QCH = 512;
constexpr int KK  = 11;
constexpr int KK2 = KK * KK;        // 121
constexpr int KKP = 128;            // padded to multiple of 32

// ---------------------------------------------------------------------------
// WMMA / TDM types
// ---------------------------------------------------------------------------
typedef __bf16 v16bf __attribute__((ext_vector_type(16)));
typedef float  v8f   __attribute__((ext_vector_type(8)));
typedef unsigned int u32x4 __attribute__((ext_vector_type(4)));
typedef int          i32x4 __attribute__((ext_vector_type(4)));
typedef int          i32x8 __attribute__((ext_vector_type(8)));

#if defined(__has_builtin)
#if __has_builtin(__builtin_amdgcn_tensor_load_to_lds)
#define HAVE_TDM 1
#endif
#endif

#ifdef HAVE_TDM
__device__ __forceinline__ void tdm_issue(const u32x4& g0, const i32x8& g1) {
    i32x4 gz = {0, 0, 0, 0};
#if __clang_major__ >= 23
    i32x8 gz8 = {0, 0, 0, 0, 0, 0, 0, 0};
    __builtin_amdgcn_tensor_load_to_lds(g0, g1, gz, gz, gz8, 0);
#else
    __builtin_amdgcn_tensor_load_to_lds(g0, g1, gz, gz, 0);
#endif
}

// 2-D fp32 tile DMA: global(tile_h rows x tile_w elems, row stride `stride`
// elems) -> LDS at lds_byte with 1-dword pad after every row
// (pad_code: 4 => pad after 32 dwords, 5 => after 64 dwords).
// Reads past tensor_w/tensor_h return ZERO (hardware OOB handling).
__device__ __forceinline__ void tdm_load_2d(uint32_t lds_byte, const float* gaddr,
                                            uint32_t tile_w, uint32_t tile_h,
                                            uint32_t tensor_w, uint32_t tensor_h,
                                            uint64_t stride, uint32_t pad_code)
{
    uint64_t ga = (uint64_t)(uintptr_t)gaddr;
    u32x4 g0;
    g0[0] = 1u;                                              // count=1 (user D#)
    g0[1] = lds_byte;                                        // lds_addr
    g0[2] = (uint32_t)ga;                                    // global_addr lo
    g0[3] = (uint32_t)((ga >> 32) & 0x01FFFFFFu) | (2u << 30); // addr[56:32] | type=2
    i32x8 g1;
    g1[0] = (int)((2u << 16) | (1u << 20) | (pad_code << 22)); // data_size=4B, pad_en, pad_amount=1dw
    g1[1] = (int)((tensor_w & 0xFFFFu) << 16);                 // tensor_dim0[15:0] @bit48
    g1[2] = (int)(((tensor_w >> 16) & 0xFFFFu) | ((tensor_h & 0xFFFFu) << 16));
    g1[3] = (int)(((tensor_h >> 16) & 0xFFFFu) | ((tile_w & 0xFFFFu) << 16));
    g1[4] = (int)(tile_h & 0xFFFFu);                           // tile_dim1 (tile_dim2=0)
    g1[5] = (int)(uint32_t)(stride & 0xFFFFFFFFu);             // tensor_dim0_stride lo32
    g1[6] = (int)(uint32_t)((stride >> 32) & 0xFFFFu);         // stride hi16
    g1[7] = 0;
    tdm_issue(g0, g1);
}
#endif

__device__ __forceinline__ void async_b128(uint32_t lds_byte, const float* gaddr) {
    asm volatile("global_load_async_to_lds_b128 %0, %1, off"
                 :: "v"(lds_byte), "v"(gaddr) : "memory");
}
__device__ __forceinline__ void wait_async() {
    asm volatile("s_wait_asynccnt 0" ::: "memory");
}

// ---------------------------------------------------------------------------
// Generic bf16-WMMA GEMM:  C[M,N] = act( A[M,K] @ B[K,N] + bias + Cadd )
//  - 64x64 tile per workgroup, 8 wave32s, each wave: 16 rows x 32 cols
//  - A/B tiles staged fp32 in LDS by the Tensor Data Mover (double-buffered,
//    overlapped with WMMA via s_wait_tensorcnt<=2), converted to bf16 while
//    building fragments.
//  - Kdim multiple of 32 (callers pad); Kb = true row count of B; TDM
//    zero-fills all out-of-tensor reads.  act: 0=none 1=sigmoid 2=tanh
// ---------------------------------------------------------------------------
#define TBM 64
#define TBN 64
#define TBK 32
#define LDA_S (TBK + 1)
#define LDB_S (TBN + 1)

__global__ __launch_bounds__(256)
void gemm_bf16_wmma(const float* __restrict__ A, int lda,
                    const float* __restrict__ Bm, int ldb,
                    float* __restrict__ C, int ldc,
                    const float* __restrict__ bias,
                    const float* __restrict__ Cadd, int ldadd,
                    int Mrows, int Ncols, int Kdim, int Kb, int act)
{
    __shared__ float As[2][TBM * LDA_S];
    __shared__ float Bs[2][TBK * LDB_S];

    const int tid  = threadIdx.x;
    const int lane = tid & 31;
    const int wid  = tid >> 5;          // 0..7
    const int wm   = wid & 3;           // 4 row-tiles
    const int wn   = wid >> 2;          // 2 col-pairs
    const int row0 = wm * 16;
    const int colA = wn * 32;
    const int blockRow = blockIdx.y * TBM;
    const int blockCol = blockIdx.x * TBN;
    const int nk = Kdim / TBK;

    v8f acc0 = {};
    v8f acc1 = {};

#ifdef HAVE_TDM
    // -------- TDM path: single wave drives the DMA, double-buffered --------
    auto issue_tiles = [&](int ki, int buf) {
        int k0 = ki * TBK;
        tdm_load_2d((uint32_t)(uintptr_t)&As[buf][0],
                    A + (size_t)blockRow * lda + k0,
                    TBK, TBM,
                    (uint32_t)(Kdim - k0), (uint32_t)(Mrows - blockRow),
                    (uint64_t)lda, 4u /* pad after 32 dwords */);
        tdm_load_2d((uint32_t)(uintptr_t)&Bs[buf][0],
                    Bm + (size_t)k0 * ldb + blockCol,
                    TBN, TBK,
                    (uint32_t)(Ncols - blockCol), (uint32_t)(Kb - k0),
                    (uint64_t)ldb, 5u /* pad after 64 dwords */);
    };
    if (tid < 32) issue_tiles(0, 0);

    for (int ki = 0; ki < nk; ++ki) {
        int cur = ki & 1;
        if (tid < 32) {
            if (ki + 1 < nk) issue_tiles(ki + 1, cur ^ 1);
            // in-order TDM completion: <=2 outstanding => current pair landed
            if (ki + 1 < nk) __builtin_amdgcn_s_wait_tensorcnt(2);
            else             __builtin_amdgcn_s_wait_tensorcnt(0);
        }
        __syncthreads();
#else
    // -------- fallback: per-lane async global->LDS copies --------
    for (int ki = 0; ki < nk; ++ki) {
        int cur = 0;
        int k0 = ki * TBK;
        #pragma unroll
        for (int i = 0; i < 2; ++i) {
            int c  = tid + 256 * i;
            int r  = c >> 3, cc = c & 7;
            int gr = blockRow + r; if (gr >= Mrows) gr = Mrows - 1;
            async_b128((uint32_t)(uintptr_t)&As[0][r * LDA_S + cc * 4],
                       A + (size_t)gr * lda + k0 + cc * 4);
        }
        #pragma unroll
        for (int i = 0; i < 2; ++i) {
            int c  = tid + 256 * i;
            int r  = c >> 4, cc = c & 15;
            int gk = k0 + r; if (gk >= Kb) gk = Kb - 1;
            async_b128((uint32_t)(uintptr_t)&Bs[0][r * LDB_S + cc * 4],
                       Bm + (size_t)gk * ldb + blockCol + cc * 4);
        }
        wait_async();
        __syncthreads();
#endif
        if ((ki + 1) * TBK < Kdim) {
            __builtin_prefetch(A + (size_t)(blockRow + (tid & 63)) * lda + (ki + 1) * TBK, 0, 1);
            __builtin_prefetch(Bm + (size_t)((ki + 1) * TBK + (tid & 31)) * ldb + blockCol, 0, 1);
        }

        // ---- A fragment (16x32 bf16), contiguous k = 16*(lane/16) + e ----
        v16bf af;
        {
            int lm = lane & 15;
            int kb = (lane < 16) ? 0 : 16;
            const float* ar = &As[cur][(row0 + lm) * LDA_S];
            #pragma unroll
            for (int e = 0; e < 16; ++e) {
                int kx = kb + e;
                af[e] = (__bf16)ar[kx];
            }
        }
        // ---- B fragments (32x16 bf16), two column tiles ----
        v16bf bf0, bf1;
        {
            int nn  = lane & 15;
            int kb2 = (lane < 16) ? 0 : 16;
            #pragma unroll
            for (int e = 0; e < 16; ++e) {
                const float* br = &Bs[cur][(kb2 + e) * LDB_S];
                bf0[e] = (__bf16)br[colA + nn];
                bf1[e] = (__bf16)br[colA + 16 + nn];
            }
        }
        acc0 = __builtin_amdgcn_wmma_f32_16x16x32_bf16(false, af, false, bf0,
                                                       (short)0, acc0, false, false);
        acc1 = __builtin_amdgcn_wmma_f32_16x16x32_bf16(false, af, false, bf1,
                                                       (short)0, acc1, false, false);
        __syncthreads();
    }

    // ---- store C (16x16 f32 layout: lane<16 -> M=r, lane>=16 -> M=r+8) ----
    const int cn = lane & 15;
    const int cm = (lane < 16) ? 0 : 8;
    #pragma unroll
    for (int r = 0; r < 8; ++r) {
        int gm = blockRow + row0 + cm + r;
        if (gm >= Mrows) continue;
        size_t rowoff = (size_t)gm * ldc;
        int gc0 = blockCol + colA + cn;
        int gc1 = gc0 + 16;
        if (gc0 < Ncols) {
            float v = acc0[r];
            if (bias) v += bias[gc0];
            if (Cadd) v += Cadd[(size_t)gm * ldadd + gc0];
            if (act == 1) v = 1.f / (1.f + __expf(-v));
            else if (act == 2) v = tanhf(v);
            C[rowoff + gc0] = v;
        }
        if (gc1 < Ncols) {
            float v = acc1[r];
            if (bias) v += bias[gc1];
            if (Cadd) v += Cadd[(size_t)gm * ldadd + gc1];
            if (act == 1) v = 1.f / (1.f + __expf(-v));
            else if (act == 2) v = tanhf(v);
            C[rowoff + gc1] = v;
        }
    }
}

// ---------------------------------------------------------------------------
// Small helper kernels
// ---------------------------------------------------------------------------
__global__ void fzero(float* p, int n) {
    int i = blockIdx.x * 256 + threadIdx.x;
    if (i < n) p[i] = 0.f;
}

// context (B,D,H,W) -> ctxT[hw*B+b][DPAD] (zero-padded K)
__global__ void ctx_transpose(const float* __restrict__ ctx, float* __restrict__ dst) {
    int idx = blockIdx.x * 256 + threadIdx.x;
    if (idx >= HWB * DPAD) return;
    int d   = idx % DPAD;
    int row = idx / DPAD;
    int b   = row & 7;
    int hw  = row >> 3;
    dst[idx] = (d < DD) ? ctx[((size_t)b * DD + d) * HWX + hw] : 0.f;
}

// Q (QCH,1,K,K) -> Qt[KKP][QCH] (zero-padded rows 121..127)
__global__ void q_transpose(const float* __restrict__ Q, float* __restrict__ dst) {
    int idx = blockIdx.x * 256 + threadIdx.x;
    if (idx >= KKP * QCH) return;
    int q  = idx & (QCH - 1);
    int kk = idx >> 9;
    dst[idx] = (kk < KK2) ? Q[(size_t)q * KK2 + kk] : 0.f;
}

// alpha_past (B,H,W) -> im2col patches P[hw*B+b][KKP], zero pad
__global__ void im2col_cover(const float* __restrict__ ap, float* __restrict__ P) {
    int idx = blockIdx.x * 256 + threadIdx.x;
    if (idx >= HWB * KKP) return;
    int kk  = idx & (KKP - 1);
    int row = idx >> 7;
    int b   = row & 7;
    int hw  = row >> 3;
    int h   = hw / WW, w = hw % WW;
    float v = 0.f;
    if (kk < KK2) {
        int i = kk / KK, j = kk % KK;
        int hh = h + i - KK / 2, ww = w + j - KK / 2;
        if (hh >= 0 && hh < HH && ww >= 0 && ww < WW)
            v = ap[((size_t)b * HH + hh) * WW + ww];
    }
    P[idx] = v;
}

// al_raw[r] = tanh(prectx + q + cov) . va + va_b   (one wave32 per row)
__global__ __launch_bounds__(256)
void attend_scores(const float* __restrict__ prectx, const float* __restrict__ cov,
                   const float* __restrict__ q, const float* __restrict__ va_w,
                   const float* __restrict__ va_b, float* __restrict__ al)
{
    int lane = threadIdx.x & 31;
    int wid  = threadIdx.x >> 5;
    int r    = blockIdx.x * 8 + wid;
    if (r >= HWB) return;
    int b = r & 7;
    const float* pr = prectx + (size_t)r * AA;
    const float* cv = cov    + (size_t)r * AA;
    const float* qr = q      + (size_t)b * AA;
    float sum = 0.f;
    #pragma unroll
    for (int j = 0; j < AA / 32; ++j) {
        int a = lane + 32 * j;
        sum += tanhf(pr[a] + qr[a] + cv[a]) * va_w[a];
    }
    #pragma unroll
    for (int off = 16; off > 0; off >>= 1) sum += __shfl_xor(sum, off, 32);
    if (lane == 0) al[r] = sum + va_b[0];
}

// global-max softmax (matches reference al.max() over ALL elements), mask,
// per-batch denom, normalize, alpha_past update, write alpha & ap outputs.
__global__ __launch_bounds__(256)
void attend_softmax(const float* __restrict__ alraw, const float* __restrict__ cmask,
                    float* __restrict__ alnorm, float* __restrict__ ap,
                    float* __restrict__ out_alpha, float* __restrict__ out_ap,
                    int eps_inside)
{
    __shared__ float red[256];
    __shared__ float denom_s[BB];
    int tid = threadIdx.x;
    float mx = -1e30f;
    for (int i = tid; i < HWB; i += 256) mx = fmaxf(mx, alraw[i]);
    red[tid] = mx; __syncthreads();
    for (int s = 128; s > 0; s >>= 1) {
        if (tid < s) red[tid] = fmaxf(red[tid], red[tid + s]);
        __syncthreads();
    }
    mx = red[0];
    if (tid < BB) denom_s[tid] = 0.f;
    __syncthreads();
    int myb = tid & 7;
    float psum = 0.f;
    for (int i = tid; i < HWB; i += 256) {       // i = hw*B + b -> b == tid&7
        int hw = i >> 3;
        float e = __expf(alraw[i] - mx) * cmask[(size_t)myb * HWX + hw];
        alnorm[(size_t)myb * HWX + hw] = e;      // stash e in (B,HW) layout
        psum += e;
    }
    atomicAdd(&denom_s[myb], psum);
    __syncthreads();
    float denom = denom_s[myb];
    for (int i = tid; i < HWB; i += 256) {
        int hw = i >> 3;
        size_t o = (size_t)myb * HWX + hw;
        float e = alnorm[o];
        float a = eps_inside ? (e / (denom + 1e-10f)) : (e / denom + 1e-10f);
        alnorm[o] = a;
        float nap = ap[o] + a;
        ap[o] = nap;
        out_alpha[o] = a;
        out_ap[o]    = nap;
    }
}

// ct[b,d] = sum_hw context[b,d,hw] * alpha[b,hw]; writes padded ws ct and outputs
__global__ __launch_bounds__(256)
void attend_ct(const float* __restrict__ ctx, const float* __restrict__ alnorm,
               float* __restrict__ ctpad, float* __restrict__ out_ctX,
               float* __restrict__ out_ct, int isP)
{
    __shared__ float alr[HWX];
    int b = blockIdx.x;
    for (int i = threadIdx.x; i < HWX; i += 256)
        alr[i] = alnorm[(size_t)b * HWX + i];
    __syncthreads();
    int d = blockIdx.y * 256 + threadIdx.x;
    if (d >= DD) return;
    const float* cp = ctx + ((size_t)b * DD + d) * HWX;
    float s = 0.f;
    for (int hw = 0; hw < HWX; ++hw) s += cp[hw] * alr[hw];
    ctpad[(size_t)b * DPAD + d] = s;
    out_ctX[(size_t)b * DD + d] = s;
    out_ct[(size_t)b * 2 * DD + (isP ? DD : 0) + d] = s;
}

// GRU blend: h' = m*( z*h + (1-z)*tanh(uhh*r + extra) ) + (1-m)*h
__global__ void gru_combine(const float* __restrict__ h_in, const float* __restrict__ z,
                            const float* __restrict__ r, const float* __restrict__ uhh,
                            const float* __restrict__ extra, const float* __restrict__ lym,
                            float* __restrict__ h_out, float* __restrict__ out_copy)
{
    int i = blockIdx.x * 256 + threadIdx.x;
    if (i >= BB * NN) return;
    int b = i >> 8;
    float m  = lym[b];
    float hv = h_in[i];
    float cand = tanhf(uhh[i] * r[i] + extra[i]);
    float hn = z[i] * hv + (1.f - z[i]) * cand;
    hn = m * hn + (1.f - m) * hv;
    h_out[i] = hn;
    if (out_copy) out_copy[i] = hn;
}

// ---------------------------------------------------------------------------
// Host orchestration
// ---------------------------------------------------------------------------
enum {
    IN_LEMB = 0, IN_REMB, IN_LYMASK, IN_CTX, IN_CMASK, IN_H0,
    P_Wyz0_w, P_Wyz0_b, P_Wyr0_w, P_Wyr0_b, P_Wyh0_w, P_Wyh0_b,
    P_Wyz_w,  P_Wyz_b,  P_Wyr_w,  P_Wyr_b,  P_Wyh_w,  P_Wyh_b,
    P_Wyz1_w, P_Wyz1_b, P_Wyr1_w, P_Wyr1_b, P_Wyh1_w, P_Wyh1_b,
    P_Uhz0, P_Uhr0, P_Uhh0, P_Uhz1, P_Uhr1, P_Uhh1, P_Uhz, P_Uhr, P_Uhh,
    P_Uhz2_w, P_Uhz2_b, P_Uhr2_w, P_Uhr2_b, P_Uhh2_w, P_Uhh2_b,
    P_UaP_w, P_UaP_b, P_UaC_w, P_UaC_b,
    P_UfP_w, P_UfP_b, P_UfC_w, P_UfC_b,
    P_vaP_w, P_vaP_b, P_vaC_w, P_vaC_b,
    P_WaP, P_WaC, P_QP, P_QC,
    P_Wcz, P_Wcr, P_Wch
};

static inline void launch_gemm(const float* A, int lda, const float* B, int ldb,
                               float* C, int ldc, const float* bias,
                               const float* Cadd, int ldadd,
                               int M, int N, int K, int Kb, int act, hipStream_t s)
{
    dim3 g((N + TBN - 1) / TBN, (M + TBM - 1) / TBM);
    gemm_bf16_wmma<<<g, dim3(256), 0, s>>>(A, lda, B, ldb, C, ldc, bias,
                                           Cadd, ldadd, M, N, K, Kb, act);
}

extern "C" void kernel_launch(void* const* d_in, const int* in_sizes, int n_in,
                              void* d_out, int out_size, void* d_ws, size_t ws_size,
                              hipStream_t stream)
{
    (void)in_sizes; (void)n_in; (void)out_size; (void)ws_size;
    const float* lemb  = (const float*)d_in[IN_LEMB];
    const float* remb  = (const float*)d_in[IN_REMB];
    const float* lym   = (const float*)d_in[IN_LYMASK];
    const float* ctx   = (const float*)d_in[IN_CTX];
    const float* cmask = (const float*)d_in[IN_CMASK];
    const float* h0in  = (const float*)d_in[IN_H0];
    auto P = [&](int i) { return (const float*)d_in[i]; };
    float* out = (float*)d_out;

    // ---- output offsets (floats, reference tuple return order) ----
    const size_t O_h2 = 0, O_h1 = 65536, O_h01 = 131072;
    const size_t O_ctC = 196608, O_ctP = 371712, O_ct = 546816;
    const size_t O_cal = 897024, O_cap = 1093632, O_pal = 1290240, O_pap = 1486848;

    // ---- workspace layout (floats) ----
    float* ws = (float*)d_ws;
    size_t o = 0;
    float* ctxT   = ws + o; o += (size_t)HWB * DPAD;     // 6144 x 704
    float* pctx   = ws + o; o += (size_t)HWB * AA;       // attend-C prectx
    float* repctx = ws + o; o += (size_t)HWB * AA;       // attend-P prectx
    float* sblz   = ws + o; o += (size_t)TT * BB * NN;
    float* sblr   = ws + o; o += (size_t)TT * BB * NN;
    float* sblh   = ws + o; o += (size_t)TT * BB * NN;
    float* sbz    = ws + o; o += (size_t)TT * BB * NN;
    float* sbr    = ws + o; o += (size_t)TT * BB * NN;
    float* sbh    = ws + o; o += (size_t)TT * BB * NN;
    float* QPt    = ws + o; o += (size_t)KKP * QCH;
    float* QCt    = ws + o; o += (size_t)KKP * QCH;
    float* WfoldP = ws + o; o += (size_t)KKP * AA;
    float* WfoldC = ws + o; o += (size_t)KKP * AA;
    float* patch  = ws + o; o += (size_t)HWB * KKP;
    float* cov    = ws + o; o += (size_t)HWB * AA;
    float* alraw  = ws + o; o += HWB;
    float* alnorm = ws + o; o += HWB;
    float* qbuf   = ws + o; o += (size_t)BB * AA;
    float* apP    = ws + o; o += HWB;
    float* apC    = ws + o; o += HWB;
    float* hS     = ws + o; o += BB * NN;   // carried state
    float* h0b    = ws + o; o += BB * NN;
    float* h01b   = ws + o; o += BB * NN;
    float* h1b    = ws + o; o += BB * NN;
    float* zb     = ws + o; o += BB * NN;
    float* rb     = ws + o; o += BB * NN;
    float* uhb    = ws + o; o += BB * NN;
    float* tzb    = ws + o; o += BB * NN;
    float* trb    = ws + o; o += BB * NN;
    float* thb    = ws + o; o += BB * NN;
    float* ctPpad = ws + o; o += (size_t)BB * DPAD;
    float* ctCpad = ws + o; o += (size_t)BB * DPAD;

    auto zero = [&](float* p, int n) {
        fzero<<<dim3((n + 255) / 256), dim3(256), 0, stream>>>(p, n);
    };

    // ---- one-time (per call) precompute ----
    zero(apP, HWB); zero(apC, HWB);
    zero(ctPpad, BB * DPAD); zero(ctCpad, BB * DPAD);
    hipMemcpyAsync(hS, h0in, sizeof(float) * BB * NN, hipMemcpyDeviceToDevice, stream);

    ctx_transpose<<<dim3((HWB * DPAD + 255) / 256), 256, 0, stream>>>(ctx, ctxT);
    q_transpose<<<dim3((KKP * QCH + 255) / 256), 256, 0, stream>>>(P(P_QP), QPt);
    q_transpose<<<dim3((KKP * QCH + 255) / 256), 256, 0, stream>>>(P(P_QC), QCt);

    // pctx / repctx : 6144 x 704 @ 704 x 512
    launch_gemm(ctxT, DPAD, P(P_UaC_w), AA, pctx,   AA, P(P_UaC_b), nullptr, 0,
                HWB, AA, DPAD, DD, 0, stream);
    launch_gemm(ctxT, DPAD, P(P_UaP_w), AA, repctx, AA, P(P_UaP_b), nullptr, 0,
                HWB, AA, DPAD, DD, 0, stream);

    // embedding projections (T*B x M @ M x N)
    launch_gemm(lemb, MM, P(P_Wyz0_w), NN, sblz, NN, P(P_Wyz0_b), nullptr, 0, TT*BB, NN, MM, MM, 0, stream);
    launch_gemm(lemb, MM, P(P_Wyr0_w), NN, sblr, NN, P(P_Wyr0_b), nullptr, 0, TT*BB, NN, MM, MM, 0, stream);
    launch_gemm(lemb, MM, P(P_Wyh0_w), NN, sblh, NN, P(P_Wyh0_b), nullptr, 0, TT*BB, NN, MM, MM, 0, stream);
    launch_gemm(remb, MM, P(P_Wyz_w),  NN, sbz,  NN, P(P_Wyz_b),  nullptr, 0, TT*BB, NN, MM, MM, 0, stream);
    launch_gemm(remb, MM, P(P_Wyr_w),  NN, sbr,  NN, P(P_Wyr_b),  nullptr, 0, TT*BB, NN, MM, MM, 0, stream);
    launch_gemm(remb, MM, P(P_Wyh_w),  NN, sbh,  NN, P(P_Wyh_b),  nullptr, 0, TT*BB, NN, MM, MM, 0, stream);

    // fold coverage conv into Uf:  Wfold[128 x A] = Qt[128 x 512] @ Uf[512 x A]
    launch_gemm(QPt, QCH, P(P_UfP_w), AA, WfoldP, AA, nullptr, nullptr, 0, KKP, AA, QCH, QCH, 0, stream);
    launch_gemm(QCt, QCH, P(P_UfC_w), AA, WfoldC, AA, nullptr, nullptr, 0, KKP, AA, QCH, QCH, 0, stream);

    // ---- recurrent steps ----
    for (int t = 0; t < TT; ++t) {
        const float* lz = sblz + (size_t)t * BB * NN;
        const float* lr = sblr + (size_t)t * BB * NN;
        const float* lh = sblh + (size_t)t * BB * NN;
        const float* rz = sbz  + (size_t)t * BB * NN;
        const float* rr = sbr  + (size_t)t * BB * NN;
        const float* rh = sbh  + (size_t)t * BB * NN;
        const float* lymt = lym + (size_t)t * BB;

        // GRU 0 : h -> h0
        launch_gemm(hS, NN, P(P_Uhz0), NN, zb,  NN, nullptr, lz, NN, BB, NN, NN, NN, 1, stream);
        launch_gemm(hS, NN, P(P_Uhr0), NN, rb,  NN, nullptr, lr, NN, BB, NN, NN, NN, 1, stream);
        launch_gemm(hS, NN, P(P_Uhh0), NN, uhb, NN, nullptr, nullptr, 0, BB, NN, NN, NN, 0, stream);
        gru_combine<<<dim3(8), 256, 0, stream>>>(hS, zb, rb, uhb, lh, lymt, h0b, nullptr);

        // attend P (eps inside)
        launch_gemm(h0b, NN, P(P_WaP), AA, qbuf, AA, nullptr, nullptr, 0, BB, AA, NN, NN, 0, stream);
        im2col_cover<<<dim3((HWB * KKP + 255) / 256), 256, 0, stream>>>(apP, patch);
        launch_gemm(patch, KKP, WfoldP, AA, cov, AA, P(P_UfP_b), nullptr, 0, HWB, AA, KKP, KKP, 0, stream);
        attend_scores<<<dim3(HWB / 8), 256, 0, stream>>>(repctx, cov, qbuf, P(P_vaP_w), P(P_vaP_b), alraw);
        attend_softmax<<<dim3(1), 256, 0, stream>>>(alraw, cmask, alnorm, apP,
                out + O_pal + (size_t)t * HWB, out + O_pap + (size_t)t * HWB, 1);
        attend_ct<<<dim3(BB, 3), 256, 0, stream>>>(ctx, alnorm, ctPpad,
                out + O_ctP + (size_t)t * BB * DD, out + O_ct + (size_t)t * BB * 2 * DD, 1);

        // GRU 1 : h0 -> h01   (ctP-conditioned)
        launch_gemm(ctPpad, DPAD, P(P_Wyz1_w), NN, tzb, NN, P(P_Wyz1_b), nullptr, 0, BB, NN, DPAD, DD, 0, stream);
        launch_gemm(ctPpad, DPAD, P(P_Wyr1_w), NN, trb, NN, P(P_Wyr1_b), nullptr, 0, BB, NN, DPAD, DD, 0, stream);
        launch_gemm(ctPpad, DPAD, P(P_Wyh1_w), NN, thb, NN, P(P_Wyh1_b), nullptr, 0, BB, NN, DPAD, DD, 0, stream);
        launch_gemm(h0b, NN, P(P_Uhz1), NN, zb,  NN, nullptr, tzb, NN, BB, NN, NN, NN, 1, stream);
        launch_gemm(h0b, NN, P(P_Uhr1), NN, rb,  NN, nullptr, trb, NN, BB, NN, NN, NN, 1, stream);
        launch_gemm(h0b, NN, P(P_Uhh1), NN, uhb, NN, nullptr, nullptr, 0, BB, NN, NN, NN, 0, stream);
        gru_combine<<<dim3(8), 256, 0, stream>>>(h0b, zb, rb, uhb, thb, lymt, h01b,
                out + O_h01 + (size_t)t * BB * NN);

        // GRU 2 : h01 -> h1
        launch_gemm(h01b, NN, P(P_Uhz), NN, zb,  NN, nullptr, rz, NN, BB, NN, NN, NN, 1, stream);
        launch_gemm(h01b, NN, P(P_Uhr), NN, rb,  NN, nullptr, rr, NN, BB, NN, NN, NN, 1, stream);
        launch_gemm(h01b, NN, P(P_Uhh), NN, uhb, NN, nullptr, nullptr, 0, BB, NN, NN, NN, 0, stream);
        gru_combine<<<dim3(8), 256, 0, stream>>>(h01b, zb, rb, uhb, rh, lymt, h1b,
                out + O_h1 + (size_t)t * BB * NN);

        // attend C (eps outside)
        launch_gemm(h1b, NN, P(P_WaC), AA, qbuf, AA, nullptr, nullptr, 0, BB, AA, NN, NN, 0, stream);
        im2col_cover<<<dim3((HWB * KKP + 255) / 256), 256, 0, stream>>>(apC, patch);
        launch_gemm(patch, KKP, WfoldC, AA, cov, AA, P(P_UfC_b), nullptr, 0, HWB, AA, KKP, KKP, 0, stream);
        attend_scores<<<dim3(HWB / 8), 256, 0, stream>>>(pctx, cov, qbuf, P(P_vaC_w), P(P_vaC_b), alraw);
        attend_softmax<<<dim3(1), 256, 0, stream>>>(alraw, cmask, alnorm, apC,
                out + O_cal + (size_t)t * HWB, out + O_cap + (size_t)t * HWB, 0);
        attend_ct<<<dim3(BB, 3), 256, 0, stream>>>(ctx, alnorm, ctCpad,
                out + O_ctC + (size_t)t * BB * DD, out + O_ct + (size_t)t * BB * 2 * DD, 0);

        // GRU 3 : h1 -> h2 (carried)
        launch_gemm(ctCpad, DPAD, P(P_Wcz), NN, tzb, NN, nullptr, nullptr, 0, BB, NN, DPAD, DD, 0, stream);
        launch_gemm(ctCpad, DPAD, P(P_Wcr), NN, trb, NN, nullptr, nullptr, 0, BB, NN, DPAD, DD, 0, stream);
        launch_gemm(ctCpad, DPAD, P(P_Wch), NN, thb, NN, nullptr, nullptr, 0, BB, NN, DPAD, DD, 0, stream);
        launch_gemm(h1b, NN, P(P_Uhz2_w), NN, zb,  NN, P(P_Uhz2_b), tzb, NN, BB, NN, NN, NN, 1, stream);
        launch_gemm(h1b, NN, P(P_Uhr2_w), NN, rb,  NN, P(P_Uhr2_b), trb, NN, BB, NN, NN, NN, 1, stream);
        launch_gemm(h1b, NN, P(P_Uhh2_w), NN, uhb, NN, P(P_Uhh2_b), nullptr, 0, BB, NN, NN, NN, 0, stream);
        gru_combine<<<dim3(8), 256, 0, stream>>>(h1b, zb, rb, uhb, thb, lymt, hS,
                out + O_h2 + (size_t)t * BB * NN);
    }
}
